// FP8Linear_49495203119489
// MI455X (gfx1250) — 8-bit matrix operand order repaired and checked by reading, compiled, not run on hardware
//
#include <hip/hip_runtime.h>
#include <stdint.h>

typedef __attribute__((ext_vector_type(16))) int   v16i;
typedef __attribute__((ext_vector_type(8)))  float v8f;

#define E4M3_MAX 448.0f

__global__ void init_ws_kernel(uint32_t* amax_bits) {
    amax_bits[0] = 0u;
    amax_bits[1] = 0u;
}

__global__ __launch_bounds__(256) void amax_kernel(const float* __restrict__ x,
                                                   size_t n,
                                                   uint32_t* __restrict__ amax_bits) {
    __shared__ float red[256];
    float m = 0.0f;
    size_t stride = (size_t)gridDim.x * blockDim.x * 4u;
    size_t i = ((size_t)blockIdx.x * blockDim.x + threadIdx.x) * 4u;
    for (size_t p = i; p < n; p += stride) {
        float4 v = *(const float4*)(x + p);
        m = fmaxf(m, fmaxf(fmaxf(fabsf(v.x), fabsf(v.y)),
                           fmaxf(fabsf(v.z), fabsf(v.w))));
    }
    red[threadIdx.x] = m;
    __syncthreads();
    for (int s = 128; s > 0; s >>= 1) {
        if ((int)threadIdx.x < s)
            red[threadIdx.x] = fmaxf(red[threadIdx.x], red[threadIdx.x + s]);
        __syncthreads();
    }
    if (threadIdx.x == 0)
        atomicMax((unsigned int*)amax_bits, __float_as_uint(red[0]));
}

__global__ void scales_kernel(const uint32_t* __restrict__ amax_bits,
                              float* __restrict__ scales) {
    float ax = __uint_as_float(amax_bits[0]);
    float aw = __uint_as_float(amax_bits[1]);
    float is = (ax > 0.0f) ? (E4M3_MAX / ax * 0.9f) : 1.0f;
    float ws = (aw > 0.0f) ? (E4M3_MAX / aw * 0.9f) : 1.0f;
    scales[0] = is;
    scales[1] = ws;
    scales[2] = 1.0f / (is * ws);
}

__device__ __forceinline__ uint8_t e4m3_byte_sw(float f) {
    float af = fabsf(f);
    uint8_t s = (f < 0.0f) ? 0x80 : 0x00;
    if (!(af > 0.0f)) return s;
    int e = (int)floorf(log2f(af));
    if (e < -6) e = -6;
    float p = exp2f((float)e);
    int m = (int)rintf(af / p * 8.0f);
    if (m >= 16) { m >>= 1; e++; }
    if (e > 8) { e = 8; m = 15; }
    int eb, mb;
    if (m < 8) { eb = 0; mb = m; }
    else       { eb = e + 7; mb = m - 8; }
    return s | (uint8_t)((eb << 3) | mb);
}

__device__ __forceinline__ uint32_t pack4_e4m3(float a, float b, float c, float d) {
#if defined(__has_builtin) && __has_builtin(__builtin_amdgcn_cvt_pk_fp8_f32)
    int v = __builtin_amdgcn_cvt_pk_fp8_f32(a, b, 0, false);
    v = __builtin_amdgcn_cvt_pk_fp8_f32(c, d, v, true);
    return (uint32_t)v;
#else
    return (uint32_t)e4m3_byte_sw(a)        |
           ((uint32_t)e4m3_byte_sw(b) << 8) |
           ((uint32_t)e4m3_byte_sw(c) << 16)|
           ((uint32_t)e4m3_byte_sw(d) << 24);
#endif
}

__global__ __launch_bounds__(256) void quant_kernel(const float* __restrict__ in,
                                                    uint8_t* __restrict__ out,
                                                    const float* __restrict__ scales,
                                                    int scale_idx, size_t n) {
    float s = scales[scale_idx];
    size_t i = ((size_t)blockIdx.x * blockDim.x + threadIdx.x) * 8u;
    if (i >= n) return;
    float4 v0 = *(const float4*)(in + i);
    float4 v1 = *(const float4*)(in + i + 4);
    float q0 = fminf(fmaxf(v0.x * s, -E4M3_MAX), E4M3_MAX);
    float q1 = fminf(fmaxf(v0.y * s, -E4M3_MAX), E4M3_MAX);
    float q2 = fminf(fmaxf(v0.z * s, -E4M3_MAX), E4M3_MAX);
    float q3 = fminf(fmaxf(v0.w * s, -E4M3_MAX), E4M3_MAX);
    float q4 = fminf(fmaxf(v1.x * s, -E4M3_MAX), E4M3_MAX);
    float q5 = fminf(fmaxf(v1.y * s, -E4M3_MAX), E4M3_MAX);
    float q6 = fminf(fmaxf(v1.z * s, -E4M3_MAX), E4M3_MAX);
    float q7 = fminf(fmaxf(v1.w * s, -E4M3_MAX), E4M3_MAX);
    uint2 packed;
    packed.x = pack4_e4m3(q0, q1, q2, q3);
    packed.y = pack4_e4m3(q4, q5, q6, q7);
    *(uint2*)(out + i) = packed;
}

__device__ __forceinline__ v16i load_b_frag(const uint8_t* __restrict__ bp) {
    v16i b;
#pragma unroll
    for (int j = 0; j < 4; ++j) {
        int4 q = *(const int4*)(bp + j * 32);
        b[4 * j]     = q.x;
        b[4 * j + 1] = q.y;
        b[4 * j + 2] = q.z;
        b[4 * j + 3] = q.w;
    }
    return b;
}

__global__ __launch_bounds__(256) void fp8_gemm_kernel(
        const uint8_t* __restrict__ x8, const uint8_t* __restrict__ w8,
        const float* __restrict__ bias, const float* __restrict__ scales,
        float* __restrict__ out, int M, int N, int K) {
    const int lane = threadIdx.x & 31;
    const int wave = threadIdx.x >> 5;
    const int lm   = lane & 15;
    const int lh   = lane >> 4;
    const int mbase = blockIdx.x * 128 + wave * 16;
    const int nbase = blockIdx.y * 64;

    v8f acc[4] = {v8f{}, v8f{}, v8f{}, v8f{}};

    const uint8_t* aRow = x8 + (size_t)(mbase + lm) * K + lh * 16;
    const uint8_t* bRow0 = w8 + (size_t)(nbase +  0 + lm) * K + lh * 16;
    const uint8_t* bRow1 = w8 + (size_t)(nbase + 16 + lm) * K + lh * 16;
    const uint8_t* bRow2 = w8 + (size_t)(nbase + 32 + lm) * K + lh * 16;
    const uint8_t* bRow3 = w8 + (size_t)(nbase + 48 + lm) * K + lh * 16;

    for (int kc = 0; kc < K; kc += 128) {
        v16i a;
#pragma unroll
        for (int j = 0; j < 8; ++j) {
            int2 t = *(const int2*)(aRow + kc + (j >> 1) * 32 + (j & 1) * 8);
            a[2 * j]     = t.x;
            a[2 * j + 1] = t.y;
        }
        v16i b0 = load_b_frag(bRow0 + kc);
        v16i b1 = load_b_frag(bRow1 + kc);
        v16i b2 = load_b_frag(bRow2 + kc);
        v16i b3 = load_b_frag(bRow3 + kc);

        if (kc + 128 < K)
            __builtin_prefetch(aRow + kc + 128, 0, 0);

        acc[0] = __builtin_amdgcn_wmma_f32_16x16x128_fp8_fp8(
            a, b0, (short)0, acc[0], false, false);
        acc[1] = __builtin_amdgcn_wmma_f32_16x16x128_fp8_fp8(
            a, b1, (short)0, acc[1], false, false);
        acc[2] = __builtin_amdgcn_wmma_f32_16x16x128_fp8_fp8(
            a, b2, (short)0, acc[2], false, false);
        acc[3] = __builtin_amdgcn_wmma_f32_16x16x128_fp8_fp8(
            a, b3, (short)0, acc[3], false, false);
    }

    const float inv = scales[2];
#pragma unroll
    for (int t = 0; t < 4; ++t) {
        int n = nbase + t * 16 + lm;
        float bn = bias[n];
#pragma unroll
        for (int v = 0; v < 8; ++v) {
            int row = mbase + v + 8 * lh;
            out[(size_t)row * N + n] = (acc[t][v] + bn) * inv;
        }
    }
}

extern "C" void kernel_launch(void* const* d_in, const int* in_sizes, int n_in,
                              void* d_out, int out_size, void* d_ws, size_t ws_size,
                              hipStream_t stream) {
    const float* x    = (const float*)d_in[0];
    const float* w    = (const float*)d_in[1];
    const float* bias = (const float*)d_in[2];
    float* out = (float*)d_out;

    const int N = in_sizes[2];
    const int K = in_sizes[1] / N;
    const int M = in_sizes[0] / K;
    const size_t nx = (size_t)M * K;
    const size_t nw = (size_t)N * K;

    uint32_t* amax_bits = (uint32_t*)d_ws;
    float*    scales    = (float*)((char*)d_ws + 16);
    uint8_t*  x8        = (uint8_t*)d_ws + 4096;
    uint8_t*  w8        = x8 + nx;

    init_ws_kernel<<<1, 1, 0, stream>>>(amax_bits);

    amax_kernel<<<1024, 256, 0, stream>>>(x, nx, amax_bits + 0);
    amax_kernel<<<64,   256, 0, stream>>>(w, nw, amax_bits + 1);

    scales_kernel<<<1, 1, 0, stream>>>(amax_bits, scales);

    quant_kernel<<<(unsigned)((nx / 8 + 255) / 256), 256, 0, stream>>>(x, x8, scales, 0, nx);
    quant_kernel<<<(unsigned)((nw / 8 + 255) / 256), 256, 0, stream>>>(w, w8, scales, 1, nw);

    dim3 grid(M / 128, N / 64);
    fp8_gemm_kernel<<<grid, 256, 0, stream>>>(x8, w8, bias, scales, out, M, N, K);
}
